// WindowAttention_21792664059909
// MI455X (gfx1250) — matrix operand order repaired and checked by reading, not compiled, not run on hardware
//
#include <hip/hip_runtime.h>
#include <stdint.h>

// ---------------------------------------------------------------------------
// CDNA5 (gfx1250) window attention: bf16 WMMA for all matmuls, f32 softmax.
// ---------------------------------------------------------------------------

typedef __bf16 bf16;
typedef __bf16 v16bf __attribute__((ext_vector_type(16)));
typedef float  v8f   __attribute__((ext_vector_type(8)));
typedef unsigned int u32x4 __attribute__((ext_vector_type(4)));

union FragU { u32x4 u4[2]; v16bf v; };   // 32 bytes = 16 bf16 = one A/B fragment

__device__ __forceinline__ v8f wmma_bf16(const FragU& a, const FragU& b, v8f c) {
  // D(16x16,f32) = A(16x32,bf16) x B(32x16,bf16) + C
  return __builtin_amdgcn_wmma_f32_16x16x32_bf16(
      /*neg_a=*/false, a.v, /*neg_b=*/false, b.v,
      /*c_mod=*/(short)0, c, /*reuse_a=*/false, /*reuse_b=*/false);
}

#define ZV8 ((v8f){0.f,0.f,0.f,0.f,0.f,0.f,0.f,0.f})

// Problem constants
#define BB 4
#define NN 4096
#define CC 1024
#define HH 16
#define DD 64
#define WW 256
#define NWIN 16

// ---------------------------------------------------------------------------
// Kernel 0: fp32 -> bf16 conversion
// ---------------------------------------------------------------------------
__global__ void k_cvt_bf16(const float* __restrict__ in, bf16* __restrict__ out, int n) {
  int i = blockIdx.x * blockDim.x + threadIdx.x;
  if (i < n) out[i] = (bf16)in[i];
}

// ---------------------------------------------------------------------------
// Kernel 1: QKV GEMM.  Y[m,nn] = sum_k x[m,k] * qkv_w[nn,k]
// M=16384, N=3072, K=1024. WG tile 128x256, 8 waves each 64x64
// (16 WMMA per 16 b128 loads per k-step => ~16.4 FLOP/byte of cache feed).
// Epilogue scatters bf16 into Q/K/V raw [B,H,N,D].
// ---------------------------------------------------------------------------
__global__ __launch_bounds__(256) void k_gemm_qkv(
    const bf16* __restrict__ A, const bf16* __restrict__ Wt,
    bf16* __restrict__ Qr, bf16* __restrict__ Kr, bf16* __restrict__ Vr) {
  const int lane = threadIdx.x & 31;
  const int hf   = lane >> 4;      // lane half (0/1)
  const int l15  = lane & 15;
  const int wv   = threadIdx.x >> 5;
  const int moff = blockIdx.y * 128 + (wv & 1) * 64;
  const int noff = blockIdx.x * 256 + (wv >> 1) * 64;

  v8f acc[4][4];
#pragma unroll
  for (int t = 0; t < 4; ++t)
#pragma unroll
    for (int c = 0; c < 4; ++c) acc[t][c] = ZV8;

  for (int k0 = 0; k0 < CC; k0 += 32) {
    FragU af[4], bfr[4];
#pragma unroll
    for (int t = 0; t < 4; ++t) {
      const bf16* p = A + (size_t)(moff + t * 16 + l15) * CC + k0 + 8 * hf;
      af[t].u4[0] = *(const u32x4*)p;          // K = k0+8h .. +7
      af[t].u4[1] = *(const u32x4*)(p + 16);   // K = k0+16+8h .. +7
    }
#pragma unroll
    for (int c = 0; c < 4; ++c) {
      const bf16* p = Wt + (size_t)(noff + c * 16 + l15) * CC + k0 + 8 * hf;
      bfr[c].u4[0] = *(const u32x4*)p;         // K = k0+8h .. +7
      bfr[c].u4[1] = *(const u32x4*)(p + 16);  // K = k0+16+8h .. +7
    }
#pragma unroll
    for (int t = 0; t < 4; ++t)
#pragma unroll
      for (int c = 0; c < 4; ++c) acc[t][c] = wmma_bf16(af[t], bfr[c], acc[t][c]);
  }

  // Scatter: col nn -> (which, h, d); row m -> (b, n)
#pragma unroll
  for (int t = 0; t < 4; ++t)
#pragma unroll
    for (int c = 0; c < 4; ++c)
#pragma unroll
      for (int r = 0; r < 8; ++r) {
        int m  = moff + t * 16 + r + 8 * hf;
        int nn = noff + c * 16 + l15;
        int which = nn >> 10, rem = nn & 1023;
        int h = rem >> 6, d = rem & 63;
        int b = m >> 12, n = m & 4095;
        bf16* dst = (which == 0) ? Qr : (which == 1) ? Kr : Vr;
        dst[(((size_t)(b * HH + h) * NN + n) << 6) + d] = (bf16)acc[t][c][r];
      }
}

// ---------------------------------------------------------------------------
// Kernel 2: RoPE in place on Q (with 1/sqrt(D) folded in) and K.
// ---------------------------------------------------------------------------
__global__ void k_rope(bf16* __restrict__ Qr, bf16* __restrict__ Kr,
                       const float* __restrict__ fr) {
  int idx = blockIdx.x * blockDim.x + threadIdx.x;   // BB*HH*NN*32 threads
  int i  = idx & 31;
  int n  = (idx >> 5) & (NN - 1);
  int bh = idx >> 17;
  float f0 = fr[(n * 32 + i) * 2 + 0];
  float f1 = fr[(n * 32 + i) * 2 + 1];
  size_t base = ((size_t)bh * NN + n) * DD + 2 * i;
  float a = (float)Qr[base], b = (float)Qr[base + 1];
  Qr[base]     = (bf16)((a * f0 - b * f1) * 0.125f);   // D^-0.5 = 0.125
  Qr[base + 1] = (bf16)((b * f0 + a * f1) * 0.125f);
  a = (float)Kr[base]; b = (float)Kr[base + 1];
  Kr[base]     = (bf16)(a * f0 - b * f1);
  Kr[base + 1] = (bf16)(b * f0 + a * f1);
}

// ---------------------------------------------------------------------------
// Kernel 3: window attention. One WG (8 waves) per (b,h,window).
// Each wave owns 32 query rows; streaming softmax over key chunks of 32.
// ---------------------------------------------------------------------------
__global__ __launch_bounds__(256) void k_win_attn(
    const bf16* __restrict__ Qr, const bf16* __restrict__ Kr,
    const bf16* __restrict__ Vr, bf16* __restrict__ attnb) {
  __shared__ bf16 sVt[DD * WW];        // V transposed: [d][key]  (32 KB)
  __shared__ bf16 sP[8][32 * 32];      // per-wave P staging     (16 KB)

  const int bid  = blockIdx.x;
  const int wwin = bid & 15;
  const int h    = (bid >> 4) & 15;
  const int b    = bid >> 8;
  const int bh   = b * HH + h;
  const int tid  = threadIdx.x;
  const int lane = tid & 31, hf = lane >> 4, l15 = lane & 15;
  const int wv   = tid >> 5;
  const size_t winbase = ((size_t)bh * NN + wwin * WW) * DD;

  // Stage V^T into LDS (coalesced global read, transposed LDS write)
  for (int e = tid; e < WW * DD; e += 256) {
    int key = e >> 6, d = e & 63;
    sVt[d * WW + key] = Vr[winbase + e];
  }
  __syncthreads();

  const int qbase = wv * 32;

  // Q fragments: constant over key chunks. qf[tile][kstep]
  FragU qf[2][2];
#pragma unroll
  for (int t = 0; t < 2; ++t)
#pragma unroll
    for (int ks = 0; ks < 2; ++ks) {
      const bf16* qp = Qr + winbase + (size_t)(qbase + t * 16 + l15) * DD + ks * 32 + 8 * hf;
      qf[t][ks].u4[0] = *(const u32x4*)qp;
      qf[t][ks].u4[1] = *(const u32x4*)(qp + 16);
    }

  v8f o[2][4];
  float mrow[2][8], lrow[2][8];
#pragma unroll
  for (int t = 0; t < 2; ++t) {
#pragma unroll
    for (int db = 0; db < 4; ++db) o[t][db] = ZV8;
#pragma unroll
    for (int r = 0; r < 8; ++r) { mrow[t][r] = -3.0e38f; lrow[t][r] = 0.f; }
  }

  for (int c0 = 0; c0 < WW; c0 += 32) {   // key chunk
    // K fragments (B of S = Q*K^T): kf[coltile][kstep]
    FragU kf[2][2];
#pragma unroll
    for (int c = 0; c < 2; ++c)
#pragma unroll
      for (int ks = 0; ks < 2; ++ks) {
        const bf16* kp = Kr + winbase + (size_t)(c0 + c * 16 + l15) * DD + ks * 32 + 8 * hf;
        kf[c][ks].u4[0] = *(const u32x4*)kp;
        kf[c][ks].u4[1] = *(const u32x4*)(kp + 16);
      }
    // V fragments (B of P*V) from transposed LDS: vf[dblock]
    FragU vf[4];
#pragma unroll
    for (int db = 0; db < 4; ++db) {
      const bf16* vp = &sVt[(db * 16 + l15) * WW + c0 + 8 * hf];
      vf[db].u4[0] = *(const u32x4*)vp;
      vf[db].u4[1] = *(const u32x4*)(vp + 16);
    }

#pragma unroll
    for (int t = 0; t < 2; ++t) {
      v8f s0 = ZV8, s1 = ZV8;
#pragma unroll
      for (int ks = 0; ks < 2; ++ks) s0 = wmma_bf16(qf[t][ks], kf[0][ks], s0);
#pragma unroll
      for (int ks = 0; ks < 2; ++ks) s1 = wmma_bf16(qf[t][ks], kf[1][ks], s1);

      // Per-row chunk max over 32 keys (16 lanes hold one row each half)
      float cm[8];
#pragma unroll
      for (int r = 0; r < 8; ++r) cm[r] = fmaxf(s0[r], s1[r]);
#pragma unroll
      for (int mask = 1; mask <= 8; mask <<= 1)
#pragma unroll
        for (int r = 0; r < 8; ++r) cm[r] = fmaxf(cm[r], __shfl_xor(cm[r], mask, 16));

      float rs[8];
#pragma unroll
      for (int r = 0; r < 8; ++r) {
        float mn = fmaxf(mrow[t][r], cm[r]);
        float sc = __expf(mrow[t][r] - mn);
        mrow[t][r] = mn;
        float p0 = __expf(s0[r] - mn);
        float p1 = __expf(s1[r] - mn);
        s0[r] = p0; s1[r] = p1;
        rs[r] = p0 + p1;
        lrow[t][r] *= sc;
#pragma unroll
        for (int db = 0; db < 4; ++db) o[t][db][r] *= sc;
      }
#pragma unroll
      for (int mask = 1; mask <= 8; mask <<= 1)
#pragma unroll
        for (int r = 0; r < 8; ++r) rs[r] += __shfl_xor(rs[r], mask, 16);
#pragma unroll
      for (int r = 0; r < 8; ++r) lrow[t][r] += rs[r];

      // Stage P (C/D layout -> LDS row-major [row][key-in-chunk])
#pragma unroll
      for (int r = 0; r < 8; ++r) {
        int rl = t * 16 + r + 8 * hf;
        sP[wv][rl * 32 + l15]      = (bf16)s0[r];
        sP[wv][rl * 32 + 16 + l15] = (bf16)s1[r];
      }
    }

    asm volatile("s_wait_dscnt 0x0" ::: "memory");  // LDS stores visible to this wave

    // Reload P in A-fragment layout, then O += P * V
    FragU pf[2];
#pragma unroll
    for (int t = 0; t < 2; ++t) {
      const bf16* pp = &sP[wv][(t * 16 + l15) * 32 + 8 * hf];
      pf[t].u4[0] = *(const u32x4*)pp;
      pf[t].u4[1] = *(const u32x4*)(pp + 16);
    }
#pragma unroll
    for (int t = 0; t < 2; ++t)
#pragma unroll
      for (int db = 0; db < 4; ++db) o[t][db] = wmma_bf16(pf[t], vf[db], o[t][db]);
  }

  // Normalize and write attn output as bf16 in [B, N, H*D] layout
#pragma unroll
  for (int t = 0; t < 2; ++t)
#pragma unroll
    for (int db = 0; db < 4; ++db)
#pragma unroll
      for (int r = 0; r < 8; ++r) {
        int row = qbase + t * 16 + r + 8 * hf;
        int n   = wwin * WW + row;
        int d   = db * 16 + l15;
        float v = o[t][db][r] / lrow[t][r];
        attnb[((size_t)(b * NN + n)) * CC + h * DD + d] = (bf16)v;
      }
}

// ---------------------------------------------------------------------------
// Kernel 4: output projection. out[m,nn] = sum_k attn[m,k]*proj_w[nn,k] + b[nn]
// M=16384, N=1024, K=1024. WG tile 128x256, 8 waves each 64x64; fp32 output.
// ---------------------------------------------------------------------------
__global__ __launch_bounds__(256) void k_gemm_proj(
    const bf16* __restrict__ A, const bf16* __restrict__ Wt,
    const float* __restrict__ bias, float* __restrict__ out) {
  const int lane = threadIdx.x & 31;
  const int hf   = lane >> 4;
  const int l15  = lane & 15;
  const int wv   = threadIdx.x >> 5;
  const int moff = blockIdx.y * 128 + (wv & 1) * 64;
  const int noff = blockIdx.x * 256 + (wv >> 1) * 64;

  v8f acc[4][4];
#pragma unroll
  for (int t = 0; t < 4; ++t)
#pragma unroll
    for (int c = 0; c < 4; ++c) acc[t][c] = ZV8;

  for (int k0 = 0; k0 < CC; k0 += 32) {
    FragU af[4], bfr[4];
#pragma unroll
    for (int t = 0; t < 4; ++t) {
      const bf16* p = A + (size_t)(moff + t * 16 + l15) * CC + k0 + 8 * hf;
      af[t].u4[0] = *(const u32x4*)p;
      af[t].u4[1] = *(const u32x4*)(p + 16);
    }
#pragma unroll
    for (int c = 0; c < 4; ++c) {
      const bf16* p = Wt + (size_t)(noff + c * 16 + l15) * CC + k0 + 8 * hf;
      bfr[c].u4[0] = *(const u32x4*)p;
      bfr[c].u4[1] = *(const u32x4*)(p + 16);
    }
#pragma unroll
    for (int t = 0; t < 4; ++t)
#pragma unroll
      for (int c = 0; c < 4; ++c) acc[t][c] = wmma_bf16(af[t], bfr[c], acc[t][c]);
  }

#pragma unroll
  for (int t = 0; t < 4; ++t)
#pragma unroll
    for (int c = 0; c < 4; ++c)
#pragma unroll
      for (int r = 0; r < 8; ++r) {
        int m  = moff + t * 16 + r + 8 * hf;
        int nn = noff + c * 16 + l15;
        out[(size_t)m * CC + nn] = acc[t][c][r] + bias[nn];
      }
}

// ---------------------------------------------------------------------------
// Launch
// ---------------------------------------------------------------------------
extern "C" void kernel_launch(void* const* d_in, const int* in_sizes, int n_in,
                              void* d_out, int out_size, void* d_ws, size_t ws_size,
                              hipStream_t stream) {
  (void)in_sizes; (void)n_in; (void)out_size; (void)ws_size;

  const float* x      = (const float*)d_in[0];   // (B,N,C)
  const float* freqs  = (const float*)d_in[1];   // (N,D/2,2)
  const float* qkv_w  = (const float*)d_in[2];   // (3C,C)
  const float* proj_w = (const float*)d_in[3];   // (C,C)
  const float* proj_b = (const float*)d_in[4];   // (C,)
  float* out = (float*)d_out;

  char* ws = (char*)d_ws;
  const size_t nX  = (size_t)BB * NN * CC;       // 16,777,216
  const size_t nWq = (size_t)3 * CC * CC;        //  3,145,728
  const size_t nWp = (size_t)CC * CC;            //  1,048,576
  const size_t nQ  = (size_t)BB * HH * NN * DD;  // 16,777,216

  bf16* xb    = (bf16*)ws;  ws += nX  * sizeof(bf16);
  bf16* wqkvb = (bf16*)ws;  ws += nWq * sizeof(bf16);
  bf16* wprjb = (bf16*)ws;  ws += nWp * sizeof(bf16);
  bf16* Qr    = (bf16*)ws;  ws += nQ  * sizeof(bf16);
  bf16* Kr    = (bf16*)ws;  ws += nQ  * sizeof(bf16);
  bf16* Vr    = (bf16*)ws;  ws += nQ  * sizeof(bf16);
  bf16* attnb = (bf16*)ws;  ws += nX  * sizeof(bf16);

  k_cvt_bf16<<<(int)((nX  + 255) / 256), 256, 0, stream>>>(x, xb, (int)nX);
  k_cvt_bf16<<<(int)((nWq + 255) / 256), 256, 0, stream>>>(qkv_w, wqkvb, (int)nWq);
  k_cvt_bf16<<<(int)((nWp + 255) / 256), 256, 0, stream>>>(proj_w, wprjb, (int)nWp);

  k_gemm_qkv<<<dim3(3 * CC / 256, BB * NN / 128), 256, 0, stream>>>(xb, wqkvb, Qr, Kr, Vr);

  k_rope<<<(BB * HH * NN * 32) / 256, 256, 0, stream>>>(Qr, Kr, freqs);

  k_win_attn<<<BB * HH * NWIN, 256, 0, stream>>>(Qr, Kr, Vr, attnb);

  k_gemm_proj<<<dim3(CC / 256, BB * NN / 128), 256, 0, stream>>>(attnb, wprjb, proj_b, out);
}
